// CrossAttention_13752485282291
// MI455X (gfx1250) — matrix operand order repaired and checked by reading, not compiled, not run on hardware
//
#include <hip/hip_runtime.h>
#include <hip/hip_bf16.h>
#include <math.h>

typedef _Float16 v16h __attribute__((ext_vector_type(16)));
typedef _Float16 half8 __attribute__((ext_vector_type(8)));
typedef float    v8f  __attribute__((ext_vector_type(8)));

#define BATCH 8
#define NTOK 2048
#define CDIM 512
#define LHALF 1024
#define HEADS 8
#define DHEAD 64
#define QKVN  1536

#define SPITCH 33   // f32 score row pitch (gcd(33,64)=1 -> conflict-free)
#define PPITCH 40   // f16 P row pitch (80B, keeps 16B segment alignment)

// ---- fragment helpers -------------------------------------------------------
__device__ __forceinline__ v16h frag2(const _Float16* p0, const _Float16* p1) {
  union { v16h v; half8 h[2]; } u;
  u.h[0] = *(const half8*)p0;
  u.h[1] = *(const half8*)p1;
  return u.v;
}

__device__ __forceinline__ v8f wmma_f16(v16h a, v16h b, v8f c) {
  return __builtin_amdgcn_wmma_f32_16x16x32_f16(false, a, false, b, (short)0, c,
                                                false, false);
}

// ---- prep kernels -----------------------------------------------------------
__global__ void __launch_bounds__(256) cvt_f16_kernel(const float* __restrict__ src,
                                                      _Float16* __restrict__ dst, int n) {
  int i = blockIdx.x * 256 + threadIdx.x;
  if (i < n) dst[i] = (_Float16)src[i];
}

// W[K][N] (f32) -> Wt[N][K] (f16)
__global__ void __launch_bounds__(256) transpose_f16_kernel(const float* __restrict__ W,
                                                            _Float16* __restrict__ Wt,
                                                            int K, int N) {
  int i = blockIdx.x * 256 + threadIdx.x;
  if (i < K * N) {
    int k = i / N, n = i % N;
    Wt[(size_t)n * K + k] = (_Float16)W[i];
  }
}

// ---- QKV GEMM: per wave a 32x64 tile of (8192 x 1536), K=512 ----------------
// A: xh rows (f16), B: Wt rows (f16, [N][K]); B frags reused across 2 A frags.
__global__ void __launch_bounds__(256) qkv_kernel(
    const _Float16* __restrict__ xh,
    const _Float16* __restrict__ Wt_rgb, const float* __restrict__ b_rgb,
    const _Float16* __restrict__ Wt_dep, const float* __restrict__ b_dep,
    _Float16* __restrict__ Qh, _Float16* __restrict__ Kh, _Float16* __restrict__ Vt) {
  const int wave = threadIdx.x >> 5;
  const int lane = threadIdx.x & 31;
  const int hi   = lane >> 4;
  const int ln   = lane & 15;
  const int s    = blockIdx.z;                 // stream: 0=rgb, 1=depth
  const int tm   = blockIdx.x * 8 + wave;      // 0..255 (32-row tiles of 8192)
  const int tn   = blockIdx.y;                 // 0..23  (64-col tiles of 1536)

  const _Float16* Wt  = s ? Wt_dep : Wt_rgb;
  const float*    bia = s ? b_dep  : b_rgb;

  const int m0 = tm * 32;
  const _Float16* arow[2];
#pragma unroll
  for (int i = 0; i < 2; ++i) {
    const int tok = m0 + i * 16 + ln;          // lane's A-matrix row
    const int bA = tok >> 10, lA = tok & 1023;
    arow[i] = xh + ((size_t)(bA * NTOK + s * LHALF + lA)) * CDIM;
  }

  v8f acc[2][4] = {{v8f{}, v8f{}, v8f{}, v8f{}}, {v8f{}, v8f{}, v8f{}, v8f{}}};
  for (int k = 0; k < CDIM; k += 32) {
    v16h a0 = frag2(arow[0] + k + 8 * hi, arow[0] + k + 16 + 8 * hi);
    v16h a1 = frag2(arow[1] + k + 8 * hi, arow[1] + k + 16 + 8 * hi);
#pragma unroll
    for (int j = 0; j < 4; ++j) {
      const int n = tn * 64 + j * 16 + ln;
      const _Float16* brow = Wt + (size_t)n * CDIM + k;
      v16h bf = frag2(brow + 8 * hi, brow + 16 + 8 * hi);
      acc[0][j] = wmma_f16(a0, bf, acc[0][j]);
      acc[1][j] = wmma_f16(a1, bf, acc[1][j]);
    }
  }

#pragma unroll
  for (int j = 0; j < 4; ++j) {
    const int n = tn * 64 + j * 16 + ln;
    const int which = n >> 9;        // 0=q 1=k 2=v
    const int c = n & 511;
    const int h = c >> 6, d = c & 63;
    const float bv = bia[n];
#pragma unroll
    for (int i = 0; i < 2; ++i) {
#pragma unroll
      for (int r = 0; r < 8; ++r) {
        const int tok = m0 + i * 16 + r + 8 * hi;
        const int b = tok >> 10, l = tok & 1023;
        const float v = acc[i][j][r] + bv;
        const size_t head = (size_t)((s * BATCH + b) * HEADS + h);
        if (which == 0)
          Qh[(head * LHALF + l) * DHEAD + d] = (_Float16)(v * 0.125f);  // D^-0.5
        else if (which == 1)
          Kh[(head * LHALF + l) * DHEAD + d] = (_Float16)v;
        else
          Vt[(head * DHEAD + d) * LHALF + l] = (_Float16)v;
      }
    }
  }
}

// ---- flash attention: one wave = 32 query rows, online softmax over 1024 keys
__global__ void __launch_bounds__(256) attn_kernel(
    const _Float16* __restrict__ Qh, const _Float16* __restrict__ Kh,
    const _Float16* __restrict__ Vt, _Float16* __restrict__ outc) {
  __shared__ __attribute__((aligned(16))) float    Sbuf[8][32 * SPITCH];
  __shared__ __attribute__((aligned(16))) _Float16 Pbuf[8][32 * PPITCH];
  __shared__ float Arow[8][32];
  __shared__ float Lrow[8][32];
  __shared__ float Mrow[8][32];

  const int wave = threadIdx.x >> 5;
  const int lane = threadIdx.x & 31;
  const int hi   = lane >> 4;
  const int ln   = lane & 15;

  const int sq  = blockIdx.z;       // query stream
  const int skv = 1 - sq;           // key/value stream
  const int bh  = blockIdx.y;
  const int b = bh >> 3, h = bh & 7;
  const int q0 = blockIdx.x * 256 + wave * 32;

  const size_t qbase = (size_t)((sq  * BATCH + b) * HEADS + h) * LHALF * DHEAD;
  const size_t kbase = (size_t)((skv * BATCH + b) * HEADS + h) * LHALF * DHEAD;
  const size_t vbase = (size_t)((skv * BATCH + b) * HEADS + h) * DHEAD * LHALF;

  // Q fragments: [query subtile i][d-step], kept resident
  v16h aq[2][2];
#pragma unroll
  for (int i = 0; i < 2; ++i) {
    const _Float16* Qp = Qh + qbase + (size_t)(q0 + i * 16 + ln) * DHEAD;
    aq[i][0] = frag2(Qp + 8 * hi,      Qp + 16 + 8 * hi);
    aq[i][1] = frag2(Qp + 32 + 8 * hi, Qp + 48 + 8 * hi);
  }

  v8f O[2][4] = {{v8f{}, v8f{}, v8f{}, v8f{}}, {v8f{}, v8f{}, v8f{}, v8f{}}};
  Lrow[wave][lane] = 0.f;
  Mrow[wave][lane] = -3.0e38f;
  float*    S = Sbuf[wave];
  _Float16* P = Pbuf[wave];

  for (int kc = 0; kc < LHALF; kc += 32) {
    // scores S[32][32]; K frags loaded once, reused by both query subtiles
#pragma unroll
    for (int j = 0; j < 2; ++j) {
      const int kp = kc + j * 16 + ln;
      const _Float16* Kp = Kh + kbase + (size_t)kp * DHEAD;
      v16h bk0 = frag2(Kp + 8 * hi,      Kp + 16 + 8 * hi);
      v16h bk1 = frag2(Kp + 32 + 8 * hi, Kp + 48 + 8 * hi);
#pragma unroll
      for (int i = 0; i < 2; ++i) {
        v8f sc = {};
        sc = wmma_f16(aq[i][0], bk0, sc);
        sc = wmma_f16(aq[i][1], bk1, sc);
#pragma unroll
        for (int r = 0; r < 8; ++r)
          S[(i * 16 + r + 8 * hi) * SPITCH + j * 16 + ln] = sc[r];
      }
    }
    __builtin_amdgcn_wave_barrier();

    // per-row online softmax: all 32 lanes, one row each (wave LDS is in-order)
    {
      const int row = lane;
      const float m_old = Mrow[wave][row];
      float cmax = -3.0e38f;
#pragma unroll
      for (int c = 0; c < 32; ++c) cmax = fmaxf(cmax, S[row * SPITCH + c]);
      const float mn    = fmaxf(m_old, cmax);
      const float alpha = __expf(m_old - mn);
      float sum = 0.f;
#pragma unroll
      for (int c = 0; c < 32; ++c) {
        const float p = __expf(S[row * SPITCH + c] - mn);
        sum += p;
        P[row * PPITCH + c] = (_Float16)p;
      }
      Lrow[wave][row] = Lrow[wave][row] * alpha + sum;
      Mrow[wave][row] = mn;
      Arow[wave][row] = alpha;
    }
    __builtin_amdgcn_wave_barrier();

    // rescale O by per-row alpha
#pragma unroll
    for (int i = 0; i < 2; ++i) {
      float al[8];
#pragma unroll
      for (int r = 0; r < 8; ++r) al[r] = Arow[wave][i * 16 + r + 8 * hi];
#pragma unroll
      for (int j = 0; j < 4; ++j)
#pragma unroll
        for (int r = 0; r < 8; ++r) O[i][j][r] *= al[r];
    }

    // P as A-fragments; V frags loaded once, reused by both P fragments
    v16h ap[2];
#pragma unroll
    for (int i = 0; i < 2; ++i) {
      const _Float16* Pr = P + (i * 16 + ln) * PPITCH;
      ap[i] = frag2(Pr + 8 * hi, Pr + 16 + 8 * hi);
    }
#pragma unroll
    for (int j = 0; j < 4; ++j) {
      const int d = j * 16 + ln;
      const _Float16* Vp = Vt + vbase + (size_t)d * LHALF + kc;
      v16h bv = frag2(Vp + 8 * hi, Vp + 16 + 8 * hi);
      O[0][j] = wmma_f16(ap[0], bv, O[0][j]);
      O[1][j] = wmma_f16(ap[1], bv, O[1][j]);
    }
  }

  // rgb_out (sq==1) -> rows [0,1024); depth_out (sq==0) -> rows [1024,2048)
  const int rowbase = (sq == 0) ? LHALF : 0;
#pragma unroll
  for (int i = 0; i < 2; ++i) {
    float li[8];
#pragma unroll
    for (int r = 0; r < 8; ++r) li[r] = 1.0f / Lrow[wave][i * 16 + r + 8 * hi];
#pragma unroll
    for (int j = 0; j < 4; ++j) {
      const int col = h * DHEAD + j * 16 + ln;
#pragma unroll
      for (int r = 0; r < 8; ++r) {
        const int qpos = q0 + i * 16 + r + 8 * hi;
        outc[((size_t)(b * NTOK + rowbase + qpos)) * CDIM + col] =
            (_Float16)(O[i][j][r] * li[r]);
      }
    }
  }
}

// ---- output projection: (16384 x 512) @ (512 x 512) + bias -> f32 ----------
__global__ void __launch_bounds__(256) proj_kernel(
    const _Float16* __restrict__ outc, const _Float16* __restrict__ Wpt,
    const float* __restrict__ bproj, float* __restrict__ out) {
  const int wave = threadIdx.x >> 5;
  const int lane = threadIdx.x & 31;
  const int hi   = lane >> 4;
  const int ln   = lane & 15;
  const int tm = blockIdx.x * 8 + wave;   // 0..511 (32-row tiles)
  const int tn = blockIdx.y;              // 0..7

  const _Float16* arow0 = outc + (size_t)(tm * 32 + ln) * CDIM;
  const _Float16* arow1 = outc + (size_t)(tm * 32 + 16 + ln) * CDIM;
  v8f acc[2][4] = {{v8f{}, v8f{}, v8f{}, v8f{}}, {v8f{}, v8f{}, v8f{}, v8f{}}};
  for (int k = 0; k < CDIM; k += 32) {
    v16h a0 = frag2(arow0 + k + 8 * hi, arow0 + k + 16 + 8 * hi);
    v16h a1 = frag2(arow1 + k + 8 * hi, arow1 + k + 16 + 8 * hi);
#pragma unroll
    for (int j = 0; j < 4; ++j) {
      const int n = tn * 64 + j * 16 + ln;
      const _Float16* brow = Wpt + (size_t)n * CDIM + k;
      v16h bf = frag2(brow + 8 * hi, brow + 16 + 8 * hi);
      acc[0][j] = wmma_f16(a0, bf, acc[0][j]);
      acc[1][j] = wmma_f16(a1, bf, acc[1][j]);
    }
  }
#pragma unroll
  for (int j = 0; j < 4; ++j) {
    const int n = tn * 64 + j * 16 + ln;
    const float bb = bproj[n];
#pragma unroll
    for (int i = 0; i < 2; ++i)
#pragma unroll
      for (int r = 0; r < 8; ++r)
        out[(size_t)(tm * 32 + i * 16 + r + 8 * hi) * CDIM + n] = acc[i][j][r] + bb;
  }
}

// ---- host launcher ----------------------------------------------------------
extern "C" void kernel_launch(void* const* d_in, const int* in_sizes, int n_in,
                              void* d_out, int out_size, void* d_ws, size_t ws_size,
                              hipStream_t stream) {
  const float* x  = (const float*)d_in[0];
  const float* Wr = (const float*)d_in[1];
  const float* br = (const float*)d_in[2];
  const float* Wd = (const float*)d_in[3];
  const float* bd = (const float*)d_in[4];
  const float* Wp = (const float*)d_in[5];
  const float* bp = (const float*)d_in[6];
  float* out = (float*)d_out;

  char* ws = (char*)d_ws;
  _Float16* xh   = (_Float16*)ws; ws += (size_t)BATCH * NTOK * CDIM * 2;
  _Float16* WtR  = (_Float16*)ws; ws += (size_t)QKVN * CDIM * 2;
  _Float16* WtD  = (_Float16*)ws; ws += (size_t)QKVN * CDIM * 2;
  _Float16* WpT  = (_Float16*)ws; ws += (size_t)CDIM * CDIM * 2;
  _Float16* Qh   = (_Float16*)ws; ws += (size_t)2 * BATCH * HEADS * LHALF * DHEAD * 2;
  _Float16* Kh   = (_Float16*)ws; ws += (size_t)2 * BATCH * HEADS * LHALF * DHEAD * 2;
  _Float16* Vt   = (_Float16*)ws; ws += (size_t)2 * BATCH * HEADS * DHEAD * LHALF * 2;
  _Float16* outc = (_Float16*)ws; ws += (size_t)BATCH * NTOK * CDIM * 2;

  const int nx = BATCH * NTOK * CDIM;
  cvt_f16_kernel<<<dim3((nx + 255) / 256), dim3(256), 0, stream>>>(x, xh, nx);

  const int nqkv = CDIM * QKVN;
  transpose_f16_kernel<<<dim3((nqkv + 255) / 256), dim3(256), 0, stream>>>(Wr, WtR, CDIM, QKVN);
  transpose_f16_kernel<<<dim3((nqkv + 255) / 256), dim3(256), 0, stream>>>(Wd, WtD, CDIM, QKVN);
  const int nproj = CDIM * CDIM;
  transpose_f16_kernel<<<dim3((nproj + 255) / 256), dim3(256), 0, stream>>>(Wp, WpT, CDIM, CDIM);

  qkv_kernel<<<dim3(32, 24, 2), dim3(256), 0, stream>>>(xh, WtR, br, WtD, bd, Qh, Kh, Vt);

  attn_kernel<<<dim3(4, BATCH * HEADS, 2), dim3(256), 0, stream>>>(Qh, Kh, Vt, outc);

  proj_kernel<<<dim3(64, 8), dim3(256), 0, stream>>>(outc, WpT, bp, out);
}
